// CustomLSTMCell_85358180041592
// MI455X (gfx1250) — matrix operand order repaired and checked by reading, not compiled, not run on hardware
//
#include <hip/hip_runtime.h>
#include <hip/hip_bf16.h>

typedef __bf16 bf16_t;
typedef __attribute__((ext_vector_type(16))) __bf16 v16bf;
typedef __attribute__((ext_vector_type(8)))  __bf16 v8bf;
typedef __attribute__((ext_vector_type(8)))  float  v8f;
typedef __attribute__((ext_vector_type(4)))  int    v4i32;

#define B_SZ   8192
#define IN_SZ  1024
#define H_SZ   1024
#define D_SZ   2048   // IN_SZ + H_SZ

#define BM 128
#define BN 64         // per-gate N tile (block computes BM x BN x 4 gates)
#define BK 32
#define LDK 48        // padded LDS row (bf16 elems): 96B rows, 32B-aligned

// ---- CDNA5 async global->LDS path (guarded so compile never fails) --------
#if defined(__has_builtin)
#  if __has_builtin(__builtin_amdgcn_global_load_async_to_lds_b128)
#    define HAVE_ASYNC_LDS 1
#  endif
#  if __has_builtin(__builtin_amdgcn_s_wait_asynccnt)
#    define WAIT_ASYNC() __builtin_amdgcn_s_wait_asynccnt(0)
#  endif
#endif
#ifndef HAVE_ASYNC_LDS
#  define HAVE_ASYNC_LDS 0
#endif
#ifndef WAIT_ASYNC
#  if HAVE_ASYNC_LDS
#    define WAIT_ASYNC() asm volatile("s_wait_asynccnt 0x0" ::: "memory")
#  else
#    define WAIT_ASYNC() do {} while (0)
#  endif
#endif

#if HAVE_ASYNC_LDS
// signature (from clang diagnostic): param0 = int4 AS1*, param1 = int4 AS3*
typedef __attribute__((address_space(1))) v4i32* g_v4p;
typedef __attribute__((address_space(3))) v4i32* l_v4p;
__device__ __forceinline__ void async_copy16(const bf16_t* g, bf16_t* l) {
    __builtin_amdgcn_global_load_async_to_lds_b128((g_v4p)g, (l_v4p)l, 0, 0);
}
#endif

// ---- branch-free activations (v_exp_f32 / v_rcp_f32) ----------------------
__device__ __forceinline__ float sigmoid_fast(float x) {
    return __builtin_amdgcn_rcpf(1.f + __expf(-x));
}
__device__ __forceinline__ float tanh_fast(float x) {
    // limits: x->+inf => 1, x->-inf => -1 ; monotone, branch-free
    return 1.f - 2.f * __builtin_amdgcn_rcpf(1.f + __expf(2.f * x));
}

// ---------------------------------------------------------------------------
// Pack [input | hidden] (f32) -> A bf16 [B, D]
// ---------------------------------------------------------------------------
__global__ void pack_A_kernel(const float* __restrict__ inp,
                              const float* __restrict__ hid,
                              bf16_t* __restrict__ A) {
    int c  = blockIdx.x * blockDim.x + threadIdx.x;   // chunk of 8 elements
    int d8 = c & (D_SZ / 8 - 1);
    int b  = c >> 8;
    const float* src = (d8 < IN_SZ / 8)
                         ? (inp + (size_t)b * IN_SZ + d8 * 8)
                         : (hid + (size_t)b * H_SZ + (d8 - IN_SZ / 8) * 8);
    float4 x0 = ((const float4*)src)[0];
    float4 x1 = ((const float4*)src)[1];
    v8bf o;
    o[0] = (bf16_t)x0.x; o[1] = (bf16_t)x0.y; o[2] = (bf16_t)x0.z; o[3] = (bf16_t)x0.w;
    o[4] = (bf16_t)x1.x; o[5] = (bf16_t)x1.y; o[6] = (bf16_t)x1.z; o[7] = (bf16_t)x1.w;
    *(v8bf*)(A + (size_t)c * 8) = o;
}

// ---------------------------------------------------------------------------
// Transpose + convert W_g [D, H] f32 -> Wt[g][H][D] bf16 (k-contiguous rows)
// ---------------------------------------------------------------------------
__global__ void pack_Wt_kernel(const float* __restrict__ Wi,
                               const float* __restrict__ Wf,
                               const float* __restrict__ Wo,
                               const float* __restrict__ Wg,
                               bf16_t* __restrict__ Wt) {
    const float* W = (blockIdx.z == 0) ? Wi
                   : (blockIdx.z == 1) ? Wf
                   : (blockIdx.z == 2) ? Wo : Wg;
    bf16_t* dst = Wt + (size_t)blockIdx.z * H_SZ * D_SZ;

    __shared__ bf16_t tile[32][33];
    int tx = threadIdx.x & 31;
    int ty = threadIdx.x >> 5;
    int d0 = blockIdx.x * 32;
    int h0 = blockIdx.y * 32;
#pragma unroll
    for (int r = 0; r < 4; ++r) {
        int dd = ty + r * 8;
        tile[dd][tx] = (bf16_t)W[(size_t)(d0 + dd) * H_SZ + h0 + tx];
    }
    __syncthreads();
#pragma unroll
    for (int r = 0; r < 4; ++r) {
        int hh = ty + r * 8;
        dst[(size_t)(h0 + hh) * D_SZ + d0 + tx] = tile[tx][hh];
    }
}

// ---------------------------------------------------------------------------
// Fused gates GEMM + LSTM epilogue. 256 threads = 8 wave32, 4(M) x 2(N).
// Each wave: 2 M-sub x 2 N-sub x 4 gates = 16 WMMA accumulators.
// Double-buffered LDS; async global->LDS when the toolchain exposes it.
// ---------------------------------------------------------------------------
__global__ __launch_bounds__(256)
void lstm_gemm_kernel(const bf16_t* __restrict__ A,   // [B, D] bf16
                      const bf16_t* __restrict__ Wt,  // [4][H][D] bf16
                      const float* __restrict__ bi, const float* __restrict__ bff,
                      const float* __restrict__ bo, const float* __restrict__ bg,
                      const float* __restrict__ cell,
                      float* __restrict__ out) {
    __shared__ __align__(32) bf16_t As[2][BM][LDK];       // 2 x 12 KB
    __shared__ __align__(32) bf16_t Ws[2][4][BN][LDK];    // 2 x 24 KB

    const int tid  = threadIdx.x;
    const int lane = tid & 31;
    const int wave = tid >> 5;
    const int wm   = wave >> 1;         // 0..3 -> M offset wm*32
    const int wn   = wave & 1;          // 0..1 -> N offset wn*32

    const int bm0 = blockIdx.y * BM;
    const int bn0 = blockIdx.x * BN;

    v8f acc[2][2][4];
#pragma unroll
    for (int mi = 0; mi < 2; ++mi)
#pragma unroll
        for (int ni = 0; ni < 2; ++ni)
#pragma unroll
            for (int g = 0; g < 4; ++g)
                acc[mi][ni][g] = (v8f){0.f, 0.f, 0.f, 0.f, 0.f, 0.f, 0.f, 0.f};

    // staging indices: A tile 128x32 bf16 = 512 x 16B chunks (2/thread);
    // W tile per gate 64x32 bf16 = 256 chunks (1/thread/gate)
    const int ar0 = tid >> 2,         ac0 = (tid & 3) * 8;
    const int ar1 = (tid + 256) >> 2, ac1 = ((tid + 256) & 3) * 8;
    const int wr  = tid >> 2,         wc  = (tid & 3) * 8;

    const int kb_a = (lane < 16) ? 0 : 16;
    const int kb_b = (lane < 16) ? 0 : 16;
    const int l15  = lane & 15;

    const bf16_t* gA0 = &A[(size_t)(bm0 + ar0) * D_SZ + ac0];
    const bf16_t* gA1 = &A[(size_t)(bm0 + ar1) * D_SZ + ac1];
    const bf16_t* gW  = &Wt[(size_t)(bn0 + wr) * D_SZ + wc];

    auto stage_tile = [&](int buf, int k0) {
#if HAVE_ASYNC_LDS
        async_copy16(gA0 + k0, &As[buf][ar0][ac0]);
        async_copy16(gA1 + k0, &As[buf][ar1][ac1]);
#pragma unroll
        for (int g = 0; g < 4; ++g)
            async_copy16(gW + (size_t)g * H_SZ * D_SZ + k0, &Ws[buf][g][wr][wc]);
#else
        *(uint4*)&As[buf][ar0][ac0] = *(const uint4*)(gA0 + k0);
        *(uint4*)&As[buf][ar1][ac1] = *(const uint4*)(gA1 + k0);
#pragma unroll
        for (int g = 0; g < 4; ++g)
            *(uint4*)&Ws[buf][g][wr][wc] =
                *(const uint4*)(gW + (size_t)g * H_SZ * D_SZ + k0);
#endif
    };

    const int NK = D_SZ / BK;           // 64 K-steps
    stage_tile(0, 0);

    for (int kt = 0; kt < NK; ++kt) {
        const int buf = kt & 1;
        WAIT_ASYNC();                   // own async copies landed (async path)
        __syncthreads();                // tile kt visible to all waves

        if (kt + 1 < NK)                // prefetch next tile into other buffer
            stage_tile(buf ^ 1, (kt + 1) * BK);

        // ---- A fragments (lanes<16: K 0-15 contiguous; lanes>=16: +16) ----
        v16bf afrag[2];
#pragma unroll
        for (int mi = 0; mi < 2; ++mi) {
            const bf16_t* ap = &As[buf][wm * 32 + mi * 16 + l15][kb_a];
            v8bf lo = *(const v8bf*)ap;
            v8bf hi = *(const v8bf*)(ap + 8);
            afrag[mi] = __builtin_shufflevector(lo, hi,
                0, 1, 2, 3, 4, 5, 6, 7, 8, 9, 10, 11, 12, 13, 14, 15);
        }

        // ---- B fragments (lanes<16: K 0-15 contiguous) + 16 WMMAs ----
#pragma unroll
        for (int ni = 0; ni < 2; ++ni) {
#pragma unroll
            for (int g = 0; g < 4; ++g) {
                v16bf bfrag = *(const v16bf*)&Ws[buf][g][wn * 32 + ni * 16 + l15][kb_b];
#pragma unroll
                for (int mi = 0; mi < 2; ++mi) {
                    acc[mi][ni][g] = __builtin_amdgcn_wmma_f32_16x16x32_bf16(
                        false, afrag[mi], false, bfrag,
                        (short)0, acc[mi][ni][g], false, false);
                }
            }
        }
        __syncthreads();                // reads done before buffer reuse
    }

    // ---- fused LSTM epilogue (branch-free activations) ----
    const int mrow = ((lane >> 4) << 3);
#pragma unroll
    for (int mi = 0; mi < 2; ++mi) {
#pragma unroll
        for (int ni = 0; ni < 2; ++ni) {
            const int col = bn0 + wn * 32 + ni * 16 + l15;
            const float bI = bi[col], bF = bff[col], bO = bo[col], bG = bg[col];
#pragma unroll
            for (int j = 0; j < 8; ++j) {
                const int row = bm0 + wm * 32 + mi * 16 + mrow + j;
                const float si = sigmoid_fast(acc[mi][ni][0][j] + bI);
                const float sf = sigmoid_fast(acc[mi][ni][1][j] + bF);
                const float so = sigmoid_fast(acc[mi][ni][2][j] + bO);
                const float tg = tanh_fast(acc[mi][ni][3][j] + bG);
                const float cold = cell[(size_t)row * H_SZ + col];
                const float cnew = sf * cold + si * tg;
                const float hnew = so * tanh_fast(cnew);
                out[(size_t)row * H_SZ + col] = hnew;                        // new_hidden
                out[(size_t)B_SZ * H_SZ + (size_t)row * H_SZ + col] = cnew;  // new_cell
            }
        }
    }
}

extern "C" void kernel_launch(void* const* d_in, const int* in_sizes, int n_in,
                              void* d_out, int out_size, void* d_ws, size_t ws_size,
                              hipStream_t stream) {
    const float* inp  = (const float*)d_in[0];
    const float* hid  = (const float*)d_in[1];
    const float* cell = (const float*)d_in[2];
    const float* Wi   = (const float*)d_in[3];
    const float* bi   = (const float*)d_in[4];
    const float* bff  = (const float*)d_in[6];
    const float* Wf   = (const float*)d_in[5];
    const float* Wo   = (const float*)d_in[7];
    const float* bo   = (const float*)d_in[8];
    const float* Wg   = (const float*)d_in[9];
    const float* bg   = (const float*)d_in[10];

    bf16_t* Abf = (bf16_t*)d_ws;                                    // 32 MB
    bf16_t* Wt  = (bf16_t*)((char*)d_ws + (size_t)B_SZ * D_SZ * 2); // 16 MB

    pack_A_kernel<<<(B_SZ * D_SZ / 8) / 256, 256, 0, stream>>>(inp, hid, Abf);

    dim3 gW(D_SZ / 32, H_SZ / 32, 4);
    pack_Wt_kernel<<<gW, 256, 0, stream>>>(Wi, Wf, Wo, Wg, Wt);

    dim3 gG(H_SZ / BN, B_SZ / BM);   // (16, 64)
    lstm_gemm_kernel<<<gG, 256, 0, stream>>>(Abf, Wt, bi, bff, bo, bg, cell,
                                             (float*)d_out);
}
